// AGFMAttentionDPLM2_70592082477175
// MI455X (gfx1250) — matrix operand order repaired and checked by reading, not compiled, not run on hardware
//
#include <hip/hip_runtime.h>
#include <hip/hip_fp16.h>

// Problem constants (B, S, H, NH, HD) = (2, 2048, 1024, 16, 64)
#define B_   2
#define S_   2048
#define H_   1024
#define NH_  16
#define HD_  64
#define HALF_ (S_ / 2)
#define M_   (B_ * S_)      // 4096 rows
#define H3_  (3 * H_)       // 3072

typedef _Float16 h16;
typedef __attribute__((ext_vector_type(16))) _Float16 v16h;
typedef __attribute__((ext_vector_type(8)))  float    v8f;

// ---------------------------------------------------------------------------
// Async global->LDS copy (CDNA5 GLOBAL_LOAD_ASYNC_TO_LDS_B128, ASYNCcnt),
// guarded so we fall back to load + ds_store if the builtin is unavailable.
// Probe result: builtin takes (v4i __device__* src, v4i LDS* dst, imm, imm)
// where v4i = int __attribute__((vector_size(16))).
// ---------------------------------------------------------------------------
#if defined(__gfx1250__) &&                                         \
    __has_builtin(__builtin_amdgcn_global_load_async_to_lds_b128) && \
    __has_builtin(__builtin_amdgcn_s_wait_asynccnt)
#define USE_ASYNC_LDS 1
typedef int v4i_ __attribute__((__vector_size__(16)));
typedef __attribute__((address_space(1))) v4i_ gvec_t;
typedef __attribute__((address_space(3))) v4i_ lvec_t;
#else
#define USE_ASYNC_LDS 0
#endif

// Copy 16 halves (32 B) from global to LDS.
__device__ __forceinline__ void lds_copy16(h16* lds, const h16* g) {
#if USE_ASYNC_LDS
  __builtin_amdgcn_global_load_async_to_lds_b128((gvec_t*)g, (lvec_t*)lds, 0, 0);
  __builtin_amdgcn_global_load_async_to_lds_b128((gvec_t*)(g + 8),
                                                 (lvec_t*)(lds + 8), 0, 0);
#else
  const uint4* s = (const uint4*)g;
  *(uint4*)lds       = s[0];
  *(uint4*)(lds + 8) = s[1];
#endif
}

__device__ __forceinline__ void lds_copy_fence() {
#if USE_ASYNC_LDS
  __builtin_amdgcn_s_wait_asynccnt(0);
#endif
}

// ---------------------------------------------------------------------------
// WMMA fragment helpers (CDNA5 wave32 layouts, CDNA5 ISA §7.12.2)
// ---------------------------------------------------------------------------

// A matrix 16x32 (MxK), 16-bit, row-major storage src[m*ld + k].
__device__ __forceinline__ v16h frag_a(const h16* __restrict__ base, int ld) {
  const int lane = threadIdx.x & 31;
  const int m = lane & 15, hh = (lane >> 4) & 1;
  const h16* p = base + m * ld + (hh << 4);
  v16h a;
#pragma unroll
  for (int j = 0; j < 8; ++j) {
    const int kb = j << 1;  // 0,2,4,...,14
    a[2 * j]     = p[kb];
    a[2 * j + 1] = p[kb + 1];
  }
  return a;
}

// B matrix 32x16 (KxN), 16-bit, from N-MAJOR storage store[n*ld + k].
__device__ __forceinline__ v16h frag_b_nmajor(const h16* __restrict__ base, int ld) {
  const int lane = threadIdx.x & 31;
  const int n = lane & 15, hh = (lane >> 4) & 1;
  const h16* p = base + n * ld + (hh << 4);
  v16h b;
#pragma unroll
  for (int e = 0; e < 16; ++e) b[e] = p[e];
  return b;
}

// B matrix 32x16 (KxN), 16-bit, from K-MAJOR storage store[k*ld + n].
__device__ __forceinline__ v16h frag_b_kmajor(const h16* __restrict__ base, int ld) {
  const int lane = threadIdx.x & 31;
  const int n = lane & 15, hh = (lane >> 4) & 1;
  const h16* p = base + (hh << 4) * ld + n;
  v16h b;
#pragma unroll
  for (int e = 0; e < 16; ++e) b[e] = p[e * ld];
  return b;
}

__device__ __forceinline__ v8f wmma16(v16h a, v16h b, v8f c) {
  return __builtin_amdgcn_wmma_f32_16x16x32_f16(false, a, false, b, (short)0, c,
                                                false, false);
}

// ---------------------------------------------------------------------------
// 1) fp32 -> fp16 conversion (weights)
// ---------------------------------------------------------------------------
__global__ void k_f32_to_f16(const float* __restrict__ src, h16* __restrict__ dst, int n) {
  int i = blockIdx.x * blockDim.x + threadIdx.x;
  const int stride = gridDim.x * blockDim.x;
  for (; i < n; i += stride) dst[i] = (h16)src[i];
}

// ---------------------------------------------------------------------------
// 2) LayerNorm + AdaLN modulation (mask = s < S/2), output f16
// ---------------------------------------------------------------------------
__global__ __launch_bounds__(256) void k_ln_mod(
    const float* __restrict__ x, const float* __restrict__ g,
    const float* __restrict__ beta,
    const float* __restrict__ shift1, const float* __restrict__ shift2,
    const float* __restrict__ scale1, const float* __restrict__ scale2,
    h16* __restrict__ out) {
  __shared__ float s1[256], s2[256];
  const int row = blockIdx.x;           // b*S + s
  const int b = row / S_, s = row % S_;
  const float* xr = x + (size_t)row * H_;
  float lsum = 0.f, lsq = 0.f;
  for (int i = threadIdx.x; i < H_; i += 256) {
    const float v = xr[i];
    lsum += v; lsq += v * v;
  }
  s1[threadIdx.x] = lsum; s2[threadIdx.x] = lsq;
  __syncthreads();
  for (int o = 128; o > 0; o >>= 1) {
    if (threadIdx.x < o) {
      s1[threadIdx.x] += s1[threadIdx.x + o];
      s2[threadIdx.x] += s2[threadIdx.x + o];
    }
    __syncthreads();
  }
  const float mu  = s1[0] * (1.0f / H_);
  const float var = s2[0] * (1.0f / H_) - mu * mu;
  const float rinv = rsqrtf(var + 1e-5f);
  const float* sc = (s < HALF_) ? scale1 : scale2;
  const float* sh = (s < HALF_) ? shift1 : shift2;
  h16* orow = out + (size_t)row * H_;
  for (int i = threadIdx.x; i < H_; i += 256) {
    float y = (xr[i] - mu) * rinv * g[i] + beta[i];
    y = y * (1.0f + sc[b * H_ + i]) + sh[b * H_ + i];
    orow[i] = (h16)y;
  }
}

// ---------------------------------------------------------------------------
// 3) Fused QKV GEMM: Y[m,n] = sum_h X[m,h] * Wcat[n,h] + bias(n)
// ---------------------------------------------------------------------------
__global__ __launch_bounds__(256) void k_gemm_qkv(
    const h16* __restrict__ X,    // M_ x H_
    const h16* __restrict__ W,    // H3_ x H_ (n-major)
    const float* __restrict__ bq, const float* __restrict__ bk,
    const float* __restrict__ bv,
    h16* __restrict__ Y) {        // M_ x H3_
  __shared__ h16 As[128][40];     // row stride 80B (16B-aligned)
  __shared__ h16 Bs[128][40];
  const int bN = blockIdx.x * 128;
  const int bM = blockIdx.y * 128;
  const int t = threadIdx.x;
  const int w = t >> 5, wm = w >> 2, wn = w & 3;
  const int lane = t & 31, ln = lane & 15, hh = (lane >> 4) & 1;

  v8f acc[4][2];
#pragma unroll
  for (int tm = 0; tm < 4; ++tm)
#pragma unroll
    for (int tn = 0; tn < 2; ++tn) acc[tm][tn] = (v8f){};

  const int lr = t >> 1;           // 0..127
  const int lc = (t & 1) << 4;     // 0 or 16

  for (int kt = 0; kt < H_; kt += 32) {
    lds_copy16(&As[lr][lc], X + (size_t)(bM + lr) * H_ + kt + lc);
    lds_copy16(&Bs[lr][lc], W + (size_t)(bN + lr) * H_ + kt + lc);
    if (kt + 32 < H_) {
      __builtin_prefetch(X + (size_t)(bM + lr) * H_ + kt + 32 + lc, 0, 1);
      __builtin_prefetch(W + (size_t)(bN + lr) * H_ + kt + 32 + lc, 0, 1);
    }
    lds_copy_fence();
    __syncthreads();

    const v16h bf0 = frag_b_nmajor(&Bs[wn * 32][0], 40);
    const v16h bf1 = frag_b_nmajor(&Bs[wn * 32 + 16][0], 40);
#pragma unroll
    for (int tm = 0; tm < 4; ++tm) {
      const v16h af = frag_a(&As[wm * 64 + tm * 16][0], 40);
      acc[tm][0] = wmma16(af, bf0, acc[tm][0]);
      acc[tm][1] = wmma16(af, bf1, acc[tm][1]);
    }
    __syncthreads();
  }

#pragma unroll
  for (int tn = 0; tn < 2; ++tn) {
    const int gn = bN + wn * 32 + tn * 16 + ln;
    const float bias = (gn < H_) ? bq[gn]
                     : (gn < 2 * H_) ? bk[gn - H_] : bv[gn - 2 * H_];
#pragma unroll
    for (int tm = 0; tm < 4; ++tm) {
#pragma unroll
      for (int r = 0; r < 8; ++r) {
        const int gm = bM + wm * 64 + tm * 16 + r + 8 * hh;
        Y[(size_t)gm * H3_ + gn] = (h16)(acc[tm][tn][r] + bias);
      }
    }
  }
}

// ---------------------------------------------------------------------------
// 4) RoPE (split halves) + (B,S,NH,HD) -> (B,NH,S,HD) transpose; q *= HD^-0.5
// ---------------------------------------------------------------------------
__global__ void k_rope_transpose(const h16* __restrict__ QKV,  // (B,S,3H)
                                 h16* __restrict__ qo, h16* __restrict__ ko,
                                 h16* __restrict__ vo) {
  const int bid = blockIdx.x;               // flat (b, nh, s)
  const int s = bid % S_;
  const int nh = (bid / S_) % NH_;
  const int b = bid / (S_ * NH_);
  const int d = threadIdx.x;                // 0..63
  const size_t src = (size_t)(b * S_ + s) * H3_ + nh * HD_ + d;
  const size_t dst = (size_t)((b * NH_ + nh) * S_ + s) * HD_ + d;

  const float q = (float)QKV[src];
  const float k = (float)QKV[src + H_];
  const float v = (float)QKV[src + 2 * H_];
  const int pos = (s < HALF_) ? s : s - HALF_;
  const int d2 = d & 31;
  const float invf = __expf(-(float)d2 * 0.28782313662425572f);  // ln(1e4)/32
  const float ang = (float)pos * invf;
  float c, sn;
  __sincosf(ang, &c, &sn);
  const float sgn = (d < 32) ? -1.0f : 1.0f;
  const float qp = (float)((d < 32) ? QKV[src + 32] : QKV[src - 32]);
  const float kp = (float)((d < 32) ? QKV[src + H_ + 32] : QKV[src + H_ - 32]);
  qo[dst] = (h16)((q * c + sgn * qp * sn) * 0.125f);  // HD^-0.5 = 1/8
  ko[dst] = (h16)(k * c + sgn * kp * sn);
  vo[dst] = (h16)v;
}

// ---------------------------------------------------------------------------
// 5) Flash attention, f16 WMMA, fp32 online softmax. 64-key chunks.
//    Grid: (B*NH, S/64). 128 threads = 4 waves; wave w owns q rows
//    [blockIdx.y*64 + 16w, +16).
// ---------------------------------------------------------------------------
__global__ __launch_bounds__(128) void k_attn(
    const h16* __restrict__ Q, const h16* __restrict__ K,
    const h16* __restrict__ V, h16* __restrict__ ctx) {  // ctx: (B,S,H)
  __shared__ h16 Ks[64][72];          // [key][d], row stride 144B
  __shared__ h16 Vs[64][72];          // [key][d]
  __shared__ h16 Ps[4][16][72];       // per-wave P tile, [m][key]
  const int bh = blockIdx.x;          // b*NH + head
  const int b = bh / NH_, head = bh % NH_;
  const int t = threadIdx.x, w = t >> 5, lane = t & 31;
  const int ln = lane & 15, hh = (lane >> 4) & 1;
  const int qRow0 = blockIdx.y * 64 + w * 16;

  const h16* qb = Q + ((size_t)bh * S_ + qRow0) * HD_;
  const v16h qa0 = frag_a(qb, HD_);        // d = 0..31
  const v16h qa1 = frag_a(qb + 32, HD_);   // d = 32..63
  const h16* Kb = K + (size_t)bh * S_ * HD_;
  const h16* Vb = V + (size_t)bh * S_ * HD_;

  v8f o0 = (v8f){}, o1 = (v8f){}, o2 = (v8f){}, o3 = (v8f){};
  float mrun[8], lrun[8];
#pragma unroll
  for (int r = 0; r < 8; ++r) { mrun[r] = -1e30f; lrun[r] = 0.f; }

  const int lr = t >> 1;          // 0..63 key row
  const int lc = (t & 1) << 5;    // 0 or 32

  for (int kc = 0; kc < S_; kc += 64) {
    __syncthreads();  // protect previous chunk's Ks/Vs consumers
    const h16* gk = Kb + (size_t)(kc + lr) * HD_ + lc;
    lds_copy16(&Ks[lr][lc], gk);
    lds_copy16(&Ks[lr][lc + 16], gk + 16);
    const h16* gv = Vb + (size_t)(kc + lr) * HD_ + lc;
    lds_copy16(&Vs[lr][lc], gv);
    lds_copy16(&Vs[lr][lc + 16], gv + 16);
    if (kc + 64 < S_) {
      __builtin_prefetch(Kb + (size_t)(kc + 64 + lr) * HD_ + lc, 0, 1);
      __builtin_prefetch(Vb + (size_t)(kc + 64 + lr) * HD_ + lc, 0, 1);
    }
    lds_copy_fence();
    __syncthreads();

    // scores: c[kt] = keys [kc+16*kt, kc+16*kt+16)
    v8f c[4];
#pragma unroll
    for (int kt = 0; kt < 4; ++kt) {
      c[kt] = (v8f){};
      c[kt] = wmma16(qa0, frag_b_nmajor(&Ks[16 * kt][0], 72), c[kt]);
      c[kt] = wmma16(qa1, frag_b_nmajor(&Ks[16 * kt][32], 72), c[kt]);
    }

    // online softmax update (row r lives in 16-lane half groups)
    float corr[8];
#pragma unroll
    for (int r = 0; r < 8; ++r) {
      float vmax = fmaxf(fmaxf(c[0][r], c[1][r]), fmaxf(c[2][r], c[3][r]));
#pragma unroll
      for (int msk = 8; msk >= 1; msk >>= 1)
        vmax = fmaxf(vmax, __shfl_xor(vmax, msk, 32));
      const float mnew = fmaxf(mrun[r], vmax);
      float psum = 0.f;
#pragma unroll
      for (int kt = 0; kt < 4; ++kt) {
        const float p = __expf(c[kt][r] - mnew);
        c[kt][r] = p;
        psum += p;
      }
#pragma unroll
      for (int msk = 8; msk >= 1; msk >>= 1) psum += __shfl_xor(psum, msk, 32);
      const float cr = __expf(mrun[r] - mnew);
      lrun[r] = lrun[r] * cr + psum;
      mrun[r] = mnew;
      corr[r] = cr;
    }
#pragma unroll
    for (int r = 0; r < 8; ++r) {
      o0[r] *= corr[r]; o1[r] *= corr[r]; o2[r] *= corr[r]; o3[r] *= corr[r];
      // C-layout (row r+8h, col ln) -> A-layout staging through LDS
#pragma unroll
      for (int kt = 0; kt < 4; ++kt)
        Ps[w][r + 8 * hh][16 * kt + ln] = (h16)c[kt][r];
    }
    // same-wave DS ops are in-order; compiler inserts s_wait_dscnt
    const v16h pa0 = frag_a(&Ps[w][0][0], 72);   // keys kc..kc+31
    const v16h pa1 = frag_a(&Ps[w][0][32], 72);  // keys kc+32..kc+63
    o0 = wmma16(pa0, frag_b_kmajor(&Vs[0][0], 72), o0);
    o0 = wmma16(pa1, frag_b_kmajor(&Vs[32][0], 72), o0);
    o1 = wmma16(pa0, frag_b_kmajor(&Vs[0][16], 72), o1);
    o1 = wmma16(pa1, frag_b_kmajor(&Vs[32][16], 72), o1);
    o2 = wmma16(pa0, frag_b_kmajor(&Vs[0][32], 72), o2);
    o2 = wmma16(pa1, frag_b_kmajor(&Vs[32][32], 72), o2);
    o3 = wmma16(pa0, frag_b_kmajor(&Vs[0][48], 72), o3);
    o3 = wmma16(pa1, frag_b_kmajor(&Vs[32][48], 72), o3);
  }

#pragma unroll
  for (int r = 0; r < 8; ++r) {
    const int row = qRow0 + r + 8 * hh;
    const float inv = 1.0f / lrun[r];
    const size_t base = (size_t)(b * S_ + row) * H_ + head * HD_;
    ctx[base + ln]      = (h16)(o0[r] * inv);
    ctx[base + 16 + ln] = (h16)(o1[r] * inv);
    ctx[base + 32 + ln] = (h16)(o2[r] * inv);
    ctx[base + 48 + ln] = (h16)(o3[r] * inv);
  }
}

// ---------------------------------------------------------------------------
// 6) Output GEMM + gated residual: out = gate(b,s)*(ctx@Wo^T + bo) + hidden
// ---------------------------------------------------------------------------
__global__ __launch_bounds__(256) void k_gemm_out(
    const h16* __restrict__ X,    // ctx, M_ x H_
    const h16* __restrict__ W,    // Wo, H_ x H_ (n-major)
    const float* __restrict__ bo,
    const float* __restrict__ gate1, const float* __restrict__ gate2,
    const float* __restrict__ resid,   // original hidden_states
    float* __restrict__ out) {
  __shared__ h16 As[128][40];
  __shared__ h16 Bs[128][40];
  const int bN = blockIdx.x * 128;
  const int bM = blockIdx.y * 128;
  const int t = threadIdx.x;
  const int w = t >> 5, wm = w >> 2, wn = w & 3;
  const int lane = t & 31, ln = lane & 15, hh = (lane >> 4) & 1;

  v8f acc[4][2];
#pragma unroll
  for (int tm = 0; tm < 4; ++tm)
#pragma unroll
    for (int tn = 0; tn < 2; ++tn) acc[tm][tn] = (v8f){};

  const int lr = t >> 1;
  const int lc = (t & 1) << 4;

  for (int kt = 0; kt < H_; kt += 32) {
    lds_copy16(&As[lr][lc], X + (size_t)(bM + lr) * H_ + kt + lc);
    lds_copy16(&Bs[lr][lc], W + (size_t)(bN + lr) * H_ + kt + lc);
    if (kt + 32 < H_) {
      __builtin_prefetch(X + (size_t)(bM + lr) * H_ + kt + 32 + lc, 0, 1);
      __builtin_prefetch(W + (size_t)(bN + lr) * H_ + kt + 32 + lc, 0, 1);
    }
    lds_copy_fence();
    __syncthreads();

    const v16h bf0 = frag_b_nmajor(&Bs[wn * 32][0], 40);
    const v16h bf1 = frag_b_nmajor(&Bs[wn * 32 + 16][0], 40);
#pragma unroll
    for (int tm = 0; tm < 4; ++tm) {
      const v16h af = frag_a(&As[wm * 64 + tm * 16][0], 40);
      acc[tm][0] = wmma16(af, bf0, acc[tm][0]);
      acc[tm][1] = wmma16(af, bf1, acc[tm][1]);
    }
    __syncthreads();
  }

#pragma unroll
  for (int tn = 0; tn < 2; ++tn) {
    const int gn = bN + wn * 32 + tn * 16 + ln;
    const float bias = bo[gn];
#pragma unroll
    for (int tm = 0; tm < 4; ++tm) {
#pragma unroll
      for (int r = 0; r < 8; ++r) {
        const int gm = bM + wm * 64 + tm * 16 + r + 8 * hh;
        const int bb = gm >> 11;            // gm / S_
        const int s = gm & (S_ - 1);
        const float g = (s < HALF_) ? gate1[bb * H_ + gn] : gate2[bb * H_ + gn];
        const float hval = acc[tm][tn][r] + bias;
        out[(size_t)gm * H_ + gn] = g * hval + resid[(size_t)gm * H_ + gn];
      }
    }
  }
}

// ---------------------------------------------------------------------------
// Host launcher
// ---------------------------------------------------------------------------
extern "C" void kernel_launch(void* const* d_in, const int* in_sizes, int n_in,
                              void* d_out, int out_size, void* d_ws,
                              size_t ws_size, hipStream_t stream) {
  (void)in_sizes; (void)n_in; (void)out_size; (void)ws_size;
  const float* hidden = (const float*)d_in[0];
  const float* Wq = (const float*)d_in[1];
  const float* bq = (const float*)d_in[2];
  const float* Wk = (const float*)d_in[3];
  const float* bk = (const float*)d_in[4];
  const float* Wv = (const float*)d_in[5];
  const float* bv = (const float*)d_in[6];
  const float* Wo = (const float*)d_in[7];
  const float* bo = (const float*)d_in[8];
  const float* ln_g = (const float*)d_in[9];
  const float* ln_b = (const float*)d_in[10];
  const float* shift1 = (const float*)d_in[11];
  const float* shift2 = (const float*)d_in[12];
  const float* scale1 = (const float*)d_in[13];
  const float* scale2 = (const float*)d_in[14];
  const float* gate1 = (const float*)d_in[15];
  const float* gate2 = (const float*)d_in[16];
  // d_in[17] = type_ids: unused by the reference computation.

  char* ws = (char*)d_ws;
  const size_t MB = (size_t)1 << 20;
  h16* Wc   = (h16*)(ws);              // 3H x H f16   (6 MB)
  h16* Wo16 = (h16*)(ws + 6 * MB);     // H x H f16    (2 MB)
  h16* xmod = (h16*)(ws + 8 * MB);     // M x H f16    (8 MB)
  h16* qkv  = (h16*)(ws + 16 * MB);    // M x 3H f16   (24 MB)
  h16* qT   = (h16*)(ws + 40 * MB);    // (B,NH,S,HD)  (8 MB)
  h16* kT   = (h16*)(ws + 48 * MB);    // (8 MB)
  h16* vT   = (h16*)(ws + 56 * MB);    // (8 MB)
  h16* ctx  = (h16*)(ws + 64 * MB);    // M x H f16    (8 MB)

  const int nW = H_ * H_;
  k_f32_to_f16<<<1024, 256, 0, stream>>>(Wq, Wc, nW);
  k_f32_to_f16<<<1024, 256, 0, stream>>>(Wk, Wc + (size_t)nW, nW);
  k_f32_to_f16<<<1024, 256, 0, stream>>>(Wv, Wc + 2 * (size_t)nW, nW);
  k_f32_to_f16<<<1024, 256, 0, stream>>>(Wo, Wo16, nW);

  k_ln_mod<<<M_, 256, 0, stream>>>(hidden, ln_g, ln_b, shift1, shift2, scale1,
                                   scale2, xmod);

  k_gemm_qkv<<<dim3(H3_ / 128, M_ / 128), 256, 0, stream>>>(xmod, Wc, bq, bk,
                                                            bv, qkv);

  k_rope_transpose<<<B_ * NH_ * S_, HD_, 0, stream>>>(qkv, qT, kT, vT);

  k_attn<<<dim3(B_ * NH_, S_ / 64), 128, 0, stream>>>(qT, kT, vT, ctx);

  k_gemm_out<<<dim3(H_ / 128, M_ / 128), 256, 0, stream>>>(
      ctx, Wo16, bo, gate1, gate2, hidden, (float*)d_out);
}
